// ObjectAttentionBlock2D_76957224009951
// MI455X (gfx1250) — matrix operand order repaired and checked by reading, not compiled, not run on hardware
//
#include <hip/hip_runtime.h>

#define B_   8
#define CIN  512
#define HW_  16384
#define CK_  256
#define CV_  256
#define KP   64     // number of proxies
#define NT   128    // pixel tile per block

typedef __attribute__((ext_vector_type(16))) __bf16 v16bf;
typedef __attribute__((ext_vector_type(8)))  __bf16 v8bf;
typedef __attribute__((ext_vector_type(8)))  float  v8f;

// ---------------- LDS layout ----------------
// qs/cs : NT x LDQ bf16  = 67584   (query, later reused for context)
// xs    : NT x LDX bf16  = 133120  (x tile; region reused after query phase:)
//   ksT  : KP x LDKT bf16 = 33792   (keym transposed: [kp][q])
//   vsT  : CV x LDVT bf16 = 36864   (value transposed: [v][kp])
//   sims : NT x LDSM f32  = 36864
//   ps   : NT x KP  bf16  = 16384
#define LDX   520
#define LDQ   264
#define LDKT  264
#define LDVT  72
#define LDSM  72

#define QS_OFF 0
#define X_OFF  67584
#define KS_OFF (X_OFF + 0)
#define VS_OFF (X_OFF + 33792)
#define SM_OFF (X_OFF + 70656)
#define PS_OFF (X_OFF + 107520)
#define SMEM_BYTES (X_OFF + 133120)   // 200704 B < 320 KB/WGP

// ---------------- WMMA fragment helpers (wave32 layouts per CDNA5 ISA 7.12.2) ----------------
// Per-lane data is contiguous in K, so every fragment is two 16-byte loads.

// A fragment 16x32 bf16 from row-major [M][ld] storage.
// lane: M = lane%16, half = lane/16; elements [half*8 .. +7] and [16+half*8 .. +7].
__device__ __forceinline__ v16bf load_a_rm(const __bf16* p, int ld, int lane) {
    const int m = lane & 15, half = lane >> 4;
    const __bf16* row = p + m * ld + (half << 3);
    v8bf lo = *(const v8bf*)(row);
    v8bf hi = *(const v8bf*)(row + 16);
    return __builtin_shufflevector(lo, hi, 0, 1, 2, 3, 4, 5, 6, 7,
                                   8, 9, 10, 11, 12, 13, 14, 15);
}

// B fragment 32x16 bf16 from row-major [N][ld] storage (transposed source).
// lane: N = lane%16, half = lane/16; elements [half*8 .. +7] and [16+half*8 .. +7].
__device__ __forceinline__ v16bf load_b_nm(const __bf16* p, int ld, int lane) {
    const int n = lane & 15, half = lane >> 4;
    const __bf16* row = p + n * ld + (half << 3);
    v8bf lo = *(const v8bf*)(row);
    v8bf hi = *(const v8bf*)(row + 16);
    return __builtin_shufflevector(lo, hi, 0, 1, 2, 3, 4, 5, 6, 7,
                                   8, 9, 10, 11, 12, 13, 14, 15);
}

__device__ __forceinline__ v8f wmma_bf16(v16bf a, v16bf b, v8f c) {
    return __builtin_amdgcn_wmma_f32_16x16x32_bf16(
        /*neg_a=*/false, a, /*neg_b=*/false, b,
        /*c_mod=*/(short)0, c, /*reuse_a=*/false, /*reuse_b=*/false);
}

// ---------------- precompute kernels ----------------

__global__ void __launch_bounds__(256)
convert_weights_kernel(const float* __restrict__ Wq, const float* __restrict__ Wo,
                       __bf16* __restrict__ wq_bf, __bf16* __restrict__ wo_bf) {
    int idx = blockIdx.x * blockDim.x + threadIdx.x;
    if (idx < CK_ * CIN) {
        wq_bf[idx] = (__bf16)Wq[idx];
    } else {
        int i2 = idx - CK_ * CIN;
        if (i2 < CIN * CV_) wo_bf[i2] = (__bf16)Wo[i2];
    }
}

// keymT[b][kp][q] = Wk[q,:]·proxy[b,:,kp] + bk[q]   (bf16, stored transposed)
// valueT[b][v][kp] = Wv[v,:]·proxy[b,:,kp] + bv[v]  (bf16, stored transposed)
__global__ void __launch_bounds__(256)
proxy_proj_kernel(const float* __restrict__ proxy,
                  const float* __restrict__ Wk, const float* __restrict__ bk,
                  const float* __restrict__ Wv, const float* __restrict__ bv,
                  __bf16* __restrict__ keymT_bf, __bf16* __restrict__ valueT_bf) {
    int idx = blockIdx.x * blockDim.x + threadIdx.x;
    if (idx < B_ * KP * CK_) {
        int b = idx / (KP * CK_), r = idx % (KP * CK_);
        int kp = r >> 8, q = r & (CK_ - 1);
        const float* pcol = proxy + (size_t)b * CIN * KP + kp;
        const float* wrow = Wk + (size_t)q * CIN;
        float acc = bk[q];
        for (int c = 0; c < CIN; ++c) acc += wrow[c] * pcol[(size_t)c * KP];
        keymT_bf[idx] = (__bf16)acc;
    } else {
        int i2 = idx - B_ * KP * CK_;
        if (i2 < B_ * CV_ * KP) {
            int b = i2 / (CV_ * KP), r = i2 % (CV_ * KP);
            int v = r >> 6, kp = r & (KP - 1);
            const float* pcol = proxy + (size_t)b * CIN * KP + kp;
            const float* wrow = Wv + (size_t)v * CIN;
            float acc = bv[v];
            for (int c = 0; c < CIN; ++c) acc += wrow[c] * pcol[(size_t)c * KP];
            valueT_bf[i2] = (__bf16)acc;
        }
    }
}

// ---------------- fused attention kernel ----------------
// grid: (HW/NT, B), block: 256 threads (8 waves)
__global__ void __launch_bounds__(256)
fused_attn_kernel(const float* __restrict__ x,
                  const float* __restrict__ bq, const float* __restrict__ bo,
                  const __bf16* __restrict__ wq_bf, const __bf16* __restrict__ wo_bf,
                  const __bf16* __restrict__ keymT_bf, const __bf16* __restrict__ valueT_bf,
                  float* __restrict__ out) {
    extern __shared__ unsigned char smem[];
    __bf16* qs   = (__bf16*)(smem + QS_OFF);
    __bf16* xs   = (__bf16*)(smem + X_OFF);
    __bf16* ksT  = (__bf16*)(smem + KS_OFF);
    __bf16* vsT  = (__bf16*)(smem + VS_OFF);
    float*  sims = (float*)(smem + SM_OFF);
    __bf16* ps   = (__bf16*)(smem + PS_OFF);
    __bf16* cs   = qs;  // context reuses the query region

    const int b   = blockIdx.y;
    const int n0  = blockIdx.x * NT;
    const int tid = threadIdx.x;
    const int lane = tid & 31, wave = tid >> 5;
    const int nc = lane & 15, half = lane >> 4;
    const v8f vzero = {};

    // ---- stage x tile: [C][N] fp32 global -> [N][C] bf16 LDS (read x exactly once)
    const float* xb = x + (size_t)b * CIN * HW_ + n0;
    for (int idx = tid; idx < CIN * NT; idx += 256) {
        int c = idx >> 7, n = idx & (NT - 1);
        xs[n * LDX + c] = (__bf16)xb[(size_t)c * HW_ + n];
    }
    __syncthreads();

    // ---- query[n][q] = sum_c xs[n][c] * Wq[q][c] + bq[q]
    // Each wave owns 2 q-column tiles; one Wq fragment feeds 8 WMMAs (8 row tiles).
    for (int ni = wave * 2; ni < wave * 2 + 2; ++ni) {
        const int q0 = ni << 4;
        v8f acc[8];
#pragma unroll
        for (int i = 0; i < 8; ++i) acc[i] = vzero;
        for (int kk = 0; kk < CIN; kk += 32) {
            v16bf bm = load_b_nm(wq_bf + (size_t)q0 * CIN + kk, CIN, lane);
#pragma unroll
            for (int mi = 0; mi < 8; ++mi) {
                v16bf a = load_a_rm(xs + (mi << 4) * LDX + kk, LDX, lane);
                acc[mi] = wmma_bf16(a, bm, acc[mi]);
            }
        }
        float bias = bq[q0 + nc];
#pragma unroll
        for (int mi = 0; mi < 8; ++mi)
#pragma unroll
            for (int j = 0; j < 8; ++j)
                qs[((mi << 4) + (half << 3) + j) * LDQ + q0 + nc] =
                    (__bf16)(acc[mi][j] + bias);
    }
    __syncthreads();

    // ---- stage keymT [KP][CK] and valueT [CV][KP] into the dead x region
    // (already transposed in global; contiguous 16-byte chunk copies)
    {
        const v8bf* kg = (const v8bf*)(keymT_bf + (size_t)b * KP * CK_);
        for (int ch = tid; ch < KP * CK_ / 8; ch += 256) {
            int kp = ch >> 5, c8 = ch & 31;
            *(v8bf*)(ksT + kp * LDKT + (c8 << 3)) = kg[ch];
        }
        const v8bf* vg = (const v8bf*)(valueT_bf + (size_t)b * CV_ * KP);
        for (int ch = tid; ch < CV_ * KP / 8; ch += 256) {
            int v = ch >> 3, k8 = ch & 7;
            *(v8bf*)(vsT + v * LDVT + (k8 << 3)) = vg[ch];
        }
    }
    __syncthreads();

    // ---- sim[n][k] = (Q·keym) * CK^-1/2
    // Waves 0-3 cover k-tiles 0-3 for row tiles 0-3; waves 4-7 for row tiles 4-7.
    {
        const int ki = wave & 3;
        const int mibase = (wave >> 2) << 2;
        const int k0 = ki << 4;
        v8f acc[4];
#pragma unroll
        for (int i = 0; i < 4; ++i) acc[i] = vzero;
        for (int kk = 0; kk < CK_; kk += 32) {
            v16bf bm = load_b_nm(ksT + k0 * LDKT + kk, LDKT, lane);
#pragma unroll
            for (int i = 0; i < 4; ++i) {
                v16bf a = load_a_rm(qs + ((mibase + i) << 4) * LDQ + kk, LDQ, lane);
                acc[i] = wmma_bf16(a, bm, acc[i]);
            }
        }
#pragma unroll
        for (int i = 0; i < 4; ++i)
#pragma unroll
            for (int j = 0; j < 8; ++j)
                sims[(((mibase + i) << 4) + (half << 3) + j) * LDSM + k0 + nc] =
                    acc[i][j] * 0.0625f;
    }
    __syncthreads();

    // ---- softmax over K=64 per pixel row; probabilities -> bf16
    if (tid < NT) {
        float* srow = sims + tid * LDSM;
        float mx = -3.0e38f;
        for (int k = 0; k < KP; ++k) mx = fmaxf(mx, srow[k]);
        float sum = 0.f;
        for (int k = 0; k < KP; ++k) { float e = __expf(srow[k] - mx); srow[k] = e; sum += e; }
        float inv = 1.0f / sum;
        __bf16* prow = ps + tid * KP;
        for (int k = 0; k < KP; ++k) prow[k] = (__bf16)(srow[k] * inv);
    }
    __syncthreads();

    // ---- context[n][v] = P · value  -> cs (reuses qs region)
    // Each wave owns 2 v-column tiles; one value fragment feeds 8 WMMAs.
    for (int vi = wave * 2; vi < wave * 2 + 2; ++vi) {
        const int v0 = vi << 4;
        v8f acc[8];
#pragma unroll
        for (int i = 0; i < 8; ++i) acc[i] = vzero;
#pragma unroll
        for (int kk = 0; kk < KP; kk += 32) {
            v16bf bm = load_b_nm(vsT + v0 * LDVT + kk, LDVT, lane);
#pragma unroll
            for (int mi = 0; mi < 8; ++mi) {
                v16bf a = load_a_rm(ps + (mi << 4) * KP + kk, KP, lane);
                acc[mi] = wmma_bf16(a, bm, acc[mi]);
            }
        }
#pragma unroll
        for (int mi = 0; mi < 8; ++mi)
#pragma unroll
            for (int j = 0; j < 8; ++j)
                cs[((mi << 4) + (half << 3) + j) * LDQ + v0 + nc] = (__bf16)acc[mi][j];
    }
    __syncthreads();

    // ---- out[o][n] = Wo · context^T + bo  (fp32 store to global)
    // Each wave owns 4 o-tiles; one Wo fragment feeds 8 WMMAs (8 pixel-col tiles).
    float* ob = out + (size_t)b * CIN * HW_ + n0;
    for (int oi = wave; oi < 32; oi += 8) {
        const int o0 = oi << 4;
        v8f acc[8];
#pragma unroll
        for (int i = 0; i < 8; ++i) acc[i] = vzero;
        for (int kk = 0; kk < CV_; kk += 32) {
            v16bf a = load_a_rm(wo_bf + (size_t)o0 * CV_ + kk, CV_, lane);
#pragma unroll
            for (int ci = 0; ci < 8; ++ci) {
                v16bf bm = load_b_nm(cs + (ci << 4) * LDQ + kk, LDQ, lane);
                acc[ci] = wmma_bf16(a, bm, acc[ci]);
            }
        }
#pragma unroll
        for (int ci = 0; ci < 8; ++ci)
#pragma unroll
            for (int j = 0; j < 8; ++j) {
                int o = o0 + (half << 3) + j;
                ob[(size_t)o * HW_ + (ci << 4) + nc] = acc[ci][j] + bo[o];
            }
    }
}

// ---------------- launch ----------------
extern "C" void kernel_launch(void* const* d_in, const int* in_sizes, int n_in,
                              void* d_out, int out_size, void* d_ws, size_t ws_size,
                              hipStream_t stream) {
    const float* x     = (const float*)d_in[0];
    const float* proxy = (const float*)d_in[1];
    const float* Wq    = (const float*)d_in[2];
    const float* bq    = (const float*)d_in[3];
    const float* Wk    = (const float*)d_in[4];
    const float* bk    = (const float*)d_in[5];
    const float* Wv    = (const float*)d_in[6];
    const float* bv    = (const float*)d_in[7];
    const float* Wo    = (const float*)d_in[8];
    const float* bo    = (const float*)d_in[9];
    float* out = (float*)d_out;

    // workspace layout (bf16): Wq | Wo | keymT | valueT  = 1 MB total
    __bf16* wq_bf     = (__bf16*)d_ws;
    __bf16* wo_bf     = wq_bf + (size_t)CK_ * CIN;
    __bf16* keymT_bf  = wo_bf + (size_t)CIN * CV_;
    __bf16* valueT_bf = keymT_bf + (size_t)B_ * KP * CK_;

    convert_weights_kernel<<<(CK_ * CIN + CIN * CV_) / 256, 256, 0, stream>>>(
        Wq, Wo, wq_bf, wo_bf);
    proxy_proj_kernel<<<(B_ * KP * CK_ + B_ * CV_ * KP) / 256, 256, 0, stream>>>(
        proxy, Wk, bk, Wv, bv, keymT_bf, valueT_bf);

    dim3 grid(HW_ / NT, B_);
    fused_attn_kernel<<<grid, 256, SMEM_BYTES, stream>>>(
        x, bq, bo, wq_bf, wo_bf, keymT_bf, valueT_bf, out);
}
